// DecoderBlock_22969485099853
// MI455X (gfx1250) — matrix operand order repaired and checked by reading, not compiled, not run on hardware
//
#include <hip/hip_runtime.h>
#include <cstdint>
#include <cstddef>

// ---------------------------------------------------------------------------
// Types for CDNA5 WMMA (wave32): v_wmma_f32_16x16x32_bf16
// ---------------------------------------------------------------------------
typedef unsigned short u16;
typedef __attribute__((ext_vector_type(16))) __bf16          v16bf;
typedef __attribute__((ext_vector_type(16))) unsigned short  v16u;
typedef __attribute__((ext_vector_type(8)))  unsigned short  v8u;    // 16B = b128
typedef __attribute__((ext_vector_type(8)))  float           v8f;
typedef __attribute__((ext_vector_type(4)))  float           v4f;
typedef __attribute__((ext_vector_type(4)))  unsigned short  v4u;

__device__ __forceinline__ u16 f2bf(float f) {
  union { float f; unsigned u; } v; v.f = f;
  unsigned r = v.u + 0x7FFFu + ((v.u >> 16) & 1u);   // round-to-nearest-even
  return (u16)(r >> 16);
}

// ---------------------------------------------------------------------------
// Vectorized f32 -> bf16 conversion (all sizes here are multiples of 4)
// ---------------------------------------------------------------------------
__global__ void k_cvt(const v4f* __restrict__ x, v4u* __restrict__ y, long n4) {
  long stride = (long)gridDim.x * blockDim.x;
  for (long i = (long)blockIdx.x * blockDim.x + threadIdx.x; i < n4; i += stride) {
    v4f a = x[i];
    v4u o;
#pragma unroll
    for (int j = 0; j < 4; ++j) o[j] = f2bf(a[j]);
    y[i] = o;
  }
}

__global__ void k_zero(float* p, int n) { int i = threadIdx.x; if (i < n) p[i] = 0.f; }

// ---------------------------------------------------------------------------
// Block reductions (256 threads)
// ---------------------------------------------------------------------------
__device__ __forceinline__ float blkRedSum(float v) {
  __shared__ float sh[256];
  int t = threadIdx.x;
  sh[t] = v; __syncthreads();
  for (int o = 128; o > 0; o >>= 1) { if (t < o) sh[t] += sh[t + o]; __syncthreads(); }
  float r = sh[0]; __syncthreads(); return r;
}
__device__ __forceinline__ float blkRedMax(float v) {
  __shared__ float sh[256];
  int t = threadIdx.x;
  sh[t] = v; __syncthreads();
  for (int o = 128; o > 0; o >>= 1) { if (t < o) sh[t] = fmaxf(sh[t], sh[t + o]); __syncthreads(); }
  float r = sh[0]; __syncthreads(); return r;
}

// ---------------------------------------------------------------------------
// Column softmax over sequence axis.  Layout: X[(h*B+b)*S + s][64].
// One block per (h,b,e) column; blockIdx.x = (h*B+b)*64 + e.
// softmax(x/dq4) over s (masked self-attn: valid iff s >= e), bf16 out.
// ---------------------------------------------------------------------------
template<bool MASKED>
__global__ void k_colsoftmax(const float* __restrict__ X, u16* __restrict__ Y, int S) {
  int  cid  = blockIdx.x;
  int  e    = cid & 63;
  long base = (long)(cid >> 6) * S * 64 + e;
  const float sc = 0.35355339059327373f;   // 1 / 64^0.25
  float m = -3.0e38f;
  for (int s = threadIdx.x; s < S; s += blockDim.x) {
    if (MASKED && s < e) continue;
    m = fmaxf(m, X[base + (long)s * 64]);
  }
  m = blkRedMax(m);
  float sum = 0.f;
  for (int s = threadIdx.x; s < S; s += blockDim.x) {
    if (MASKED && s < e) continue;
    sum += __expf(sc * (X[base + (long)s * 64] - m));
  }
  sum = blkRedSum(sum);
  float inv = 1.f / sum;
  for (int s = threadIdx.x; s < S; s += blockDim.x) {
    float p = (MASKED && s < e) ? 0.f : __expf(sc * (X[base + (long)s * 64] - m)) * inv;
    Y[base + (long)s * 64] = f2bf(p);
  }
}

// ---------------------------------------------------------------------------
// LayerNorm over 1M elements per batch sample (C,S,D) = (1,2048,512)
// ---------------------------------------------------------------------------
__global__ void k_lnstats(const v4f* __restrict__ x, float* __restrict__ stats, int perB4) {
  int b = blockIdx.y;
  const v4f* xb = x + (long)b * perB4;
  float s = 0.f, s2 = 0.f;
  for (long i = (long)blockIdx.x * blockDim.x + threadIdx.x; i < perB4;
       i += (long)gridDim.x * blockDim.x) {
    v4f v = xb[i];
#pragma unroll
    for (int j = 0; j < 4; ++j) { s += v[j]; s2 += v[j] * v[j]; }
  }
  s  = blkRedSum(s);
  s2 = blkRedSum(s2);
  if (threadIdx.x == 0) { atomicAdd(&stats[2 * b], s); atomicAdd(&stats[2 * b + 1], s2); }
}

__global__ void k_lnapply(const v4f* __restrict__ x, const float* __restrict__ stats,
                          const v4f* __restrict__ g, const v4f* __restrict__ beta,
                          v4f* __restrict__ of, v4u* __restrict__ ob,
                          int perB4, long total4) {
  const float invc = 1.f / 1048576.f;     // elements per batch sample
  long stride = (long)gridDim.x * blockDim.x;
  for (long i = (long)blockIdx.x * blockDim.x + threadIdx.x; i < total4; i += stride) {
    int b = (int)(i / perB4), r = (int)(i % perB4);
    float mu  = stats[2 * b] * invc;
    float var = stats[2 * b + 1] * invc - mu * mu;
    float rs  = rsqrtf(var + 1e-5f);
    v4f v = (x[i] - mu) * rs * g[r] + beta[r];
    if (of) of[i] = v;
    if (ob) {
      v4u o;
#pragma unroll
      for (int j = 0; j < 4; ++j) o[j] = f2bf(v[j]);
      ob[i] = o;
    }
  }
}

// ---------------------------------------------------------------------------
// Generic tiled bf16 WMMA GEMM:  C = op(A) * B  [+bias] [+resid] [Swish]
// Block tile 64x64, BK=32, 128 threads = 4 waves; wave w owns rows w*16..+15.
// NN path: b128 global loads, register-staged software pipeline, prefetch.
// TA=true reads A transposed: elem(m,k) = A[k*lda + m]  (for K^T * V; cold path).
// Fragment packing per CDNA5 ISA 7.12.2:
//  A 16x32 bf16 : lane = M row (l&15), K-group = l>>4;
//                 VGPR j holds K = ((j<4)?0:16) + kg*8 + (j&3)*2, +1
//  B 32x16 bf16 : lane = N col (l&15), K-group = l>>4; same K order as A
//  C/D 16x16 f32: VGPR r, lane l -> M = r + 8*(l>>4), N = l&15
// ---------------------------------------------------------------------------
template<bool TA, int ACT>
__global__ __launch_bounds__(128) void k_gemm(
    const u16* __restrict__ A, const u16* __restrict__ B,
    const float* __restrict__ bias, long sBias,
    const float* __restrict__ resid,
    float* __restrict__ Cf, u16* __restrict__ Cb,
    int K, int lda, int ldb, int ldc,
    long sA, long sB, long sCf, long sCb)
{
  __shared__ __attribute__((aligned(16))) u16 As[64][40];  // 80B rows: 16B-aligned, odd word stride
  __shared__ __attribute__((aligned(16))) u16 Bs[32][72];  // 144B rows: 16B-aligned
  const int tid  = threadIdx.x;
  const int wave = tid >> 5, lane = tid & 31;
  const int z    = blockIdx.z;
  const long aBase = (long)z * sA, bBase = (long)z * sB;
  const int m0 = blockIdx.y * 64, n0 = blockIdx.x * 64;

  v8f z8 = {0.f, 0.f, 0.f, 0.f, 0.f, 0.f, 0.f, 0.f};
  v8f acc[4] = {z8, z8, z8, z8};

  auto compute = [&]() {
    const int row = lane & 15, kg = lane >> 4;
    v16u af;
#pragma unroll
    for (int j = 0; j < 8; ++j) {
      int kk = ((j < 4) ? 0 : 16) + kg * 8 + (j & 3) * 2;
      af[2 * j]     = As[wave * 16 + row][kk];
      af[2 * j + 1] = As[wave * 16 + row][kk + 1];
    }
    v16bf abf = __builtin_bit_cast(v16bf, af);
#pragma unroll
    for (int t = 0; t < 4; ++t) {
      v16u bfu;
#pragma unroll
      for (int j = 0; j < 8; ++j) { bfu[j] = Bs[kg * 8 + j][t * 16 + row]; bfu[8 + j] = Bs[16 + kg * 8 + j][t * 16 + row]; }
      acc[t] = __builtin_amdgcn_wmma_f32_16x16x32_bf16(
          false, abf, false, __builtin_bit_cast(v16bf, bfu), (short)0, acc[t], false, false);
    }
  };

  if constexpr (TA) {
    // cold path (tiny TN GEMMs): scalar A gather, vector B loads
    for (int k0 = 0; k0 < K; k0 += 32) {
      for (int i = tid; i < 64 * 32; i += 128) {       // coalesced over m
        int kk = i >> 6, m = i & 63;
        As[m][kk] = A[aBase + (long)(k0 + kk) * lda + (m0 + m)];
      }
#pragma unroll
      for (int u = 0; u < 2; ++u) {
        int i = tid + u * 128;                          // 256 x b128 = 32x64 tile
        *(v8u*)&Bs[i >> 3][(i & 7) * 8] =
            *(const v8u*)&B[bBase + (long)(k0 + (i >> 3)) * ldb + n0 + (i & 7) * 8];
      }
      __syncthreads();
      compute();
      __syncthreads();
    }
  } else {
    // hot path: b128 everywhere + register-staged pipeline + prefetch
    const int nt = K >> 5;
    v8u ar[2], br[2];
#pragma unroll
    for (int u = 0; u < 2; ++u) {
      int i = tid + u * 128;
      ar[u] = *(const v8u*)&A[aBase + (long)(m0 + (i >> 2)) * lda + (i & 3) * 8];
      br[u] = *(const v8u*)&B[bBase + (long)(i >> 3) * ldb + n0 + (i & 7) * 8];
    }
    for (int kt = 0; kt < nt; ++kt) {
#pragma unroll
      for (int u = 0; u < 2; ++u) {
        int i = tid + u * 128;
        *(v8u*)&As[i >> 2][(i & 3) * 8] = ar[u];
        *(v8u*)&Bs[i >> 3][(i & 7) * 8] = br[u];
      }
      __syncthreads();
      if (kt + 1 < nt) {                                // next tile -> regs (overlaps WMMA)
        int k0 = (kt + 1) * 32;
#pragma unroll
        for (int u = 0; u < 2; ++u) {
          int i = tid + u * 128;
          ar[u] = *(const v8u*)&A[aBase + (long)(m0 + (i >> 2)) * lda + k0 + (i & 3) * 8];
          br[u] = *(const v8u*)&B[bBase + (long)(k0 + (i >> 3)) * ldb + n0 + (i & 7) * 8];
        }
      }
      if (kt + 2 < nt) {                                // global_prefetch_b8, 2 tiles ahead
        int k0 = (kt + 2) * 32;
        __builtin_prefetch(&A[aBase + (long)(m0 + (tid >> 2)) * lda + k0 + (tid & 3) * 8], 0, 0);
        __builtin_prefetch(&B[bBase + (long)(k0 + (tid >> 3)) * ldb + n0 + (tid & 7) * 8], 0, 0);
      }
      compute();
      __syncthreads();
    }
  }

  // epilogue
  const int nl = lane & 15, mh = lane >> 4;
  const float* biasZ = bias ? bias + (long)z * sBias : nullptr;
#pragma unroll
  for (int t = 0; t < 4; ++t) {
    int gn = n0 + t * 16 + nl;
#pragma unroll
    for (int r = 0; r < 8; ++r) {
      int gm = m0 + wave * 16 + r + 8 * mh;
      float v = acc[t][r];
      if (biasZ) v += biasZ[gn];
      if (resid) v += resid[(long)gm * ldc + gn];
      if (ACT == 1) v = v * (1.f / (1.f + __expf(-v)));   // Swish
      long ci = (long)gm * ldc + gn;
      if (Cf) Cf[(long)z * sCf + ci] = v;
      if (Cb) Cb[(long)z * sCb + ci] = f2bf(v);
    }
  }
}

// ---------------------------------------------------------------------------
// Host side
// ---------------------------------------------------------------------------
static void gemm(hipStream_t st, bool ta, int act,
                 const u16* A, const u16* B, const float* bias, long sBias,
                 const float* resid, float* Cf, u16* Cb,
                 int M, int N, int K, int lda, int ldb, int ldc,
                 long sA, long sB, long sCf, long sCb, int batch) {
  dim3 g(N / 64, M / 64, batch), blk(128);
  if (ta)
    k_gemm<true, 0><<<g, blk, 0, st>>>(A, B, bias, sBias, resid, Cf, Cb, K, lda, ldb, ldc, sA, sB, sCf, sCb);
  else if (act == 1)
    k_gemm<false, 1><<<g, blk, 0, st>>>(A, B, bias, sBias, resid, Cf, Cb, K, lda, ldb, ldc, sA, sB, sCf, sCb);
  else
    k_gemm<false, 0><<<g, blk, 0, st>>>(A, B, bias, sBias, resid, Cf, Cb, K, lda, ldb, ldc, sA, sB, sCf, sCb);
}

static void cvt(hipStream_t st, const float* src, u16* dst, long n) {
  long n4 = n >> 2;
  long nb = (n4 + 255) / 256; if (nb > 4096) nb = 4096;
  k_cvt<<<(int)nb, 256, 0, st>>>((const v4f*)src, (v4u*)dst, n4);
}

// Workspace layout (bytes)
static constexpr size_t OFF_YB    = 0;                         // y bf16     16384*512*2
static constexpr size_t OFF_MB    = OFF_YB + 16777216;         // mem bf16   8192*512*2
static constexpr size_t OFF_W     = OFF_MB + 8388608;          // 8 attn weights bf16 (131072 B each)
static constexpr size_t OFF_WF1   = OFF_W + 8 * 131072;        // fused E1@D1 bf16 512*1024*2
static constexpr size_t OFF_WF2   = OFF_WF1 + 1048576;         // fused E2@D2 bf16
static constexpr size_t OFF_ACTA  = OFF_WF2 + 1048576;         // f32 activation 16384*512*4
static constexpr size_t OFF_STATS = OFF_ACTA + 33554432;       // 16 floats
static constexpr size_t OFF_AR    = OFF_STATS + 256;           // phase-overlaid arena (33.7 MB)
// arena: attention phase
static constexpr size_t AR_QF  = 0;                            // f32 2*16384*64
static constexpr size_t AR_KF  = AR_QF + 8388608;
static constexpr size_t AR_QB  = AR_KF + 8388608;              // bf16
static constexpr size_t AR_KB  = AR_QB + 4194304;
static constexpr size_t AR_VB  = AR_KB + 4194304;
static constexpr size_t AR_BMB = AR_VB + 4194304;              // bf16 16*64*64
static constexpr size_t AR_ZCB = AR_BMB + 131072;              // bf16 16384*128
// arena: weight-fusion phase (reuses same space, runs first)
static constexpr size_t AR_E1B = 0, AR_D1B = 2097152, AR_E2B = 6291456, AR_D2B = 10485760;
// arena: FFN phase
static constexpr size_t AR_T1B = 0;                            // bf16 16384*1024

extern "C" void kernel_launch(void* const* d_in, const int* in_sizes, int n_in,
                              void* d_out, int out_size, void* d_ws, size_t ws_size,
                              hipStream_t stream) {
  (void)in_sizes; (void)n_in; (void)out_size; (void)ws_size;
  const float* mem   = (const float*)d_in[0];
  const float* yin   = (const float*)d_in[1];
  const float* Wq_sa = (const float*)d_in[2];  const float* bq_sa = (const float*)d_in[3];
  const float* Wk_sa = (const float*)d_in[4];  const float* bk_sa = (const float*)d_in[5];
  const float* Wv_sa = (const float*)d_in[6];  const float* bv_sa = (const float*)d_in[7];
  const float* Wo_sa = (const float*)d_in[8];  const float* bo_sa = (const float*)d_in[9];
  const float* Wq_x  = (const float*)d_in[10]; const float* bq_x  = (const float*)d_in[11];
  const float* Wk_x  = (const float*)d_in[12]; const float* bk_x  = (const float*)d_in[13];
  const float* Wv_x  = (const float*)d_in[14]; const float* bv_x  = (const float*)d_in[15];
  const float* Wo_x  = (const float*)d_in[16]; const float* bo_x  = (const float*)d_in[17];
  const float* E1 = (const float*)d_in[18]; const float* D1 = (const float*)d_in[19];
  const float* E2 = (const float*)d_in[20]; const float* D2 = (const float*)d_in[21];
  const float* g1 = (const float*)d_in[22]; const float* b1 = (const float*)d_in[23];
  const float* g2 = (const float*)d_in[24]; const float* b2 = (const float*)d_in[25];
  const float* g3 = (const float*)d_in[26]; const float* b3 = (const float*)d_in[27];
  float* out = (float*)d_out;
  char*  wsb = (char*)d_ws;

  u16* YB   = (u16*)(wsb + OFF_YB);
  u16* MB   = (u16*)(wsb + OFF_MB);
  u16* WQSA = (u16*)(wsb + OFF_W + 0 * 131072);
  u16* WKSA = (u16*)(wsb + OFF_W + 1 * 131072);
  u16* WVSA = (u16*)(wsb + OFF_W + 2 * 131072);
  u16* WOSA = (u16*)(wsb + OFF_W + 3 * 131072);
  u16* WQX  = (u16*)(wsb + OFF_W + 4 * 131072);
  u16* WKX  = (u16*)(wsb + OFF_W + 5 * 131072);
  u16* WVX  = (u16*)(wsb + OFF_W + 6 * 131072);
  u16* WOX  = (u16*)(wsb + OFF_W + 7 * 131072);
  u16* WF1B = (u16*)(wsb + OFF_WF1);
  u16* WF2B = (u16*)(wsb + OFF_WF2);
  float* ACTA  = (float*)(wsb + OFF_ACTA);
  float* STATS = (float*)(wsb + OFF_STATS);
  char*  AR = wsb + OFF_AR;
  float* QF = (float*)(AR + AR_QF); float* KF = (float*)(AR + AR_KF);
  u16* QB = (u16*)(AR + AR_QB); u16* KB = (u16*)(AR + AR_KB); u16* VB = (u16*)(AR + AR_VB);
  u16* BMB = (u16*)(AR + AR_BMB); u16* ZCB = (u16*)(AR + AR_ZCB);
  u16* E1B = (u16*)(AR + AR_E1B); u16* D1B = (u16*)(AR + AR_D1B);
  u16* E2B = (u16*)(AR + AR_E2B); u16* D2B = (u16*)(AR + AR_D2B);
  u16* T1B = (u16*)(AR + AR_T1B);

  // ---- Phase W: convert + fuse LFFN weights:  Wf1 = E1@D1, Wf2 = E2@D2 ----
  cvt(stream, E1, E1B, 512L * 2048); cvt(stream, D1, D1B, 2048L * 1024);
  cvt(stream, E2, E2B, 1024L * 2048); cvt(stream, D2, D2B, 2048L * 512);
  gemm(stream, false, 0, E1B, D1B, nullptr, 0, nullptr, nullptr, WF1B,
       512, 1024, 2048, 2048, 1024, 1024, 0, 0, 0, 0, 1);
  gemm(stream, false, 0, E2B, D2B, nullptr, 0, nullptr, nullptr, WF2B,
       1024, 512, 2048, 2048, 512, 512, 0, 0, 0, 0, 1);

  // ---- Converts: activations + attention weights ----
  cvt(stream, yin, YB, 16384L * 512);
  cvt(stream, mem, MB, 8192L * 512);
  cvt(stream, Wq_sa, WQSA, 65536); cvt(stream, Wk_sa, WKSA, 65536);
  cvt(stream, Wv_sa, WVSA, 65536); cvt(stream, Wo_sa, WOSA, 65536);
  cvt(stream, Wq_x,  WQX,  65536); cvt(stream, Wk_x,  WKX,  65536);
  cvt(stream, Wv_x,  WVX,  65536); cvt(stream, Wo_x,  WOX,  65536);

  // ---- Self-attention (masked LightAttention) ----
  gemm(stream, false, 0, YB, WQSA, bq_sa, 64, nullptr, QF, nullptr,
       16384, 64, 512, 512, 64, 64, 0, 32768, 1048576, 0, 2);
  gemm(stream, false, 0, YB, WKSA, bk_sa, 64, nullptr, KF, nullptr,
       16384, 64, 512, 512, 64, 64, 0, 32768, 1048576, 0, 2);
  gemm(stream, false, 0, YB, WVSA, bv_sa, 64, nullptr, nullptr, VB,
       16384, 64, 512, 512, 64, 64, 0, 32768, 0, 1048576, 2);
  k_colsoftmax<true ><<<1024, 256, 0, stream>>>(QF, QB, 2048);
  k_colsoftmax<false><<<1024, 256, 0, stream>>>(KF, KB, 2048);
  // Bm = softmax(K^T) @ V : TN GEMM, 16 batches of (64 x 2048)x(2048 x 64)
  gemm(stream, true, 0, KB, VB, nullptr, 0, nullptr, nullptr, BMB,
       64, 64, 2048, 64, 64, 64, 131072, 131072, 0, 4096, 16);
  // Z = A @ Bm, written head-major into Zc (ldc=128, col offset h*64)
  for (int h = 0; h < 2; ++h)
    gemm(stream, false, 0, QB + (long)h * 1048576, BMB + h * 32768, nullptr, 0,
         nullptr, nullptr, ZCB + h * 64,
         2048, 64, 64, 64, 64, 128, 131072, 4096, 0, 262144, 8);
  // out_attn = Zc @ Wo + bo + y   -> ACTA
  gemm(stream, false, 0, ZCB, WOSA, bo_sa, 0, yin, ACTA, nullptr,
       16384, 512, 128, 128, 512, 512, 0, 0, 0, 0, 1);
  // LN1 -> out (f32 residual stream) + YB (bf16 for next GEMMs)
  k_zero<<<1, 32, 0, stream>>>(STATS, 16);
  k_lnstats<<<dim3(64, 8), 256, 0, stream>>>((const v4f*)ACTA, STATS, 262144);
  k_lnapply<<<4096, 256, 0, stream>>>((const v4f*)ACTA, STATS, (const v4f*)g1, (const v4f*)b1,
                                      (v4f*)out, (v4u*)YB, 262144, 2097152L);

  // ---- Cross-attention (unmasked), K/V from mem ----
  gemm(stream, false, 0, YB, WQX, bq_x, 64, nullptr, QF, nullptr,
       16384, 64, 512, 512, 64, 64, 0, 32768, 1048576, 0, 2);
  gemm(stream, false, 0, MB, WKX, bk_x, 64, nullptr, KF, nullptr,
       8192, 64, 512, 512, 64, 64, 0, 32768, 524288, 0, 2);
  gemm(stream, false, 0, MB, WVX, bv_x, 64, nullptr, nullptr, VB,
       8192, 64, 512, 512, 64, 64, 0, 32768, 0, 524288, 2);
  k_colsoftmax<false><<<1024, 256, 0, stream>>>(QF, QB, 2048);
  k_colsoftmax<false><<<1024, 256, 0, stream>>>(KF, KB, 1024);
  gemm(stream, true, 0, KB, VB, nullptr, 0, nullptr, nullptr, BMB,
       64, 64, 1024, 64, 64, 64, 65536, 65536, 0, 4096, 16);
  for (int h = 0; h < 2; ++h)
    gemm(stream, false, 0, QB + (long)h * 1048576, BMB + h * 32768, nullptr, 0,
         nullptr, nullptr, ZCB + h * 64,
         2048, 64, 64, 64, 64, 128, 131072, 4096, 0, 262144, 8);
  gemm(stream, false, 0, ZCB, WOX, bo_x, 0, out /*resid=y1*/, ACTA, nullptr,
       16384, 512, 128, 128, 512, 512, 0, 0, 0, 0, 1);
  k_zero<<<1, 32, 0, stream>>>(STATS, 16);
  k_lnstats<<<dim3(64, 8), 256, 0, stream>>>((const v4f*)ACTA, STATS, 262144);
  k_lnapply<<<4096, 256, 0, stream>>>((const v4f*)ACTA, STATS, (const v4f*)g2, (const v4f*)b2,
                                      (v4f*)out, (v4u*)YB, 262144, 2097152L);

  // ---- LFFN with pre-fused weights + Swish ----
  gemm(stream, false, 1, YB, WF1B, nullptr, 0, nullptr, nullptr, T1B,
       16384, 1024, 512, 512, 1024, 1024, 0, 0, 0, 0, 1);
  gemm(stream, false, 0, T1B, WF2B, nullptr, 0, out /*resid=y2*/, ACTA, nullptr,
       16384, 512, 1024, 1024, 512, 512, 0, 0, 0, 0, 1);
  k_zero<<<1, 32, 0, stream>>>(STATS, 16);
  k_lnstats<<<dim3(64, 8), 256, 0, stream>>>((const v4f*)ACTA, STATS, 262144);
  k_lnapply<<<4096, 256, 0, stream>>>((const v4f*)ACTA, STATS, (const v4f*)g3, (const v4f*)b3,
                                      (v4f*)out, nullptr, 262144, 2097152L);
}
